// Edge_NRI_42666205118920
// MI455X (gfx1250) — matrix operand order repaired and checked by reading, not compiled, not run on hardware
//
#include <hip/hip_runtime.h>
#include <math.h>

typedef __attribute__((ext_vector_type(16))) _Float16 v16h;
typedef __attribute__((ext_vector_type(8)))  _Float16 v8h;
typedef __attribute__((ext_vector_type(4)))  _Float16 v4h;
typedef __attribute__((ext_vector_type(8)))  float    v8f;

#define NROWS   262144      // K*N*N
#define DDIM    128
#define NTILES  16384       // NROWS/16

// ---- LDS layout (bytes) ----
#define OFF_WN2E   0         // [128][256] f16 = 65536
#define OFF_WSE1   65536     // [64][128]  f16 = 16384
#define OFF_WSE2   81920     // [128][64]  f16 = 16384
#define OFF_WEV1   98304     // [64][128]  f16 = 16384
#define OFF_BIAS   114688    // 449 floats, padded to 2048
#define PW_BASE    116736
// per-wave region
#define PW_S16     0         // 128 f16 sender row (256B)
#define PW_R16     256       // 16x128 f16 receivers (4096B)
#define PW_ALN     4352      // 16x128 f16 layernormed edges (4096B)
#define PW_AH      8448      // 16x64  f16 h (2048B)
#define PW_AZ      10496     // 16x128 f16 z (4096B)
#define PW_EFN     14592     // 16x128 f32 efn, then z in-place (8192B)
#define PW_STRIDE  22784
#define SMEM_BYTES (PW_BASE + 8*PW_STRIDE)   // 299008 B

__device__ __forceinline__ void lds_fence() {
    asm volatile("s_wait_dscnt 0" ::: "memory");
}

// hardware tanh (CDNA5 trans op v_tanh_f32)
__device__ __forceinline__ float htanh(float x) {
#if __has_builtin(__builtin_amdgcn_tanhf)
    return __builtin_amdgcn_tanhf(x);
#elif __has_builtin(__builtin_amdgcn_tanh_f32)
    return __builtin_amdgcn_tanh_f32(x);
#else
    float r; asm("v_tanh_f32 %0, %1" : "=v"(r) : "v"(x)); return r;
#endif
}

__device__ __forceinline__ float gelu_f(float x) {
    float u = 0.7978845608f * (x + 0.044715f * x * x * x);
    return 0.5f * x * (1.0f + htanh(u));
}

// A fragment: 16x32 f16, lane<16 row=lane, K chunks [k0..k0+7] & [k0+16..k0+23];
// lane>=16 same row, chunks shifted by +8. 'rowp' already points at this lane's row.
__device__ __forceinline__ v16h ldsA(const _Float16* rowp, int k0, int hi) {
    union { v16h v; v8h h[2]; } u;
    int o = k0 + (hi ? 8 : 0);
    u.h[0] = *(const v8h*)(rowp + o);
    u.h[1] = *(const v8h*)(rowp + o + 16);
    return u.v;
}

// B fragment: 32x16 f16 from transposed weights wt[N][Kt]:
// lane<16 = column n, K chunks [k0..k0+7] & [k0+16..k0+23]; lane>=16 chunks shifted by +8.
__device__ __forceinline__ v16h ldsB(const _Float16* wt, int Kt, int n, int k0, int hi) {
    const _Float16* p = wt + n * Kt + k0 + (hi ? 8 : 0);
    union { v16h v; v8h h[2]; } u;
    u.h[0] = *(const v8h*)(p);
    u.h[1] = *(const v8h*)(p + 16);
    return u.v;
}

__device__ __forceinline__ v8f wmma16(v16h a, v16h b, v8f c) {
    return __builtin_amdgcn_wmma_f32_16x16x32_f16(false, a, false, b, (short)0, c, false, false);
}

__device__ __forceinline__ v4h pack4(float4 f) {
    v4h h; h[0] = (_Float16)f.x; h[1] = (_Float16)f.y;
           h[2] = (_Float16)f.z; h[3] = (_Float16)f.w; return h;
}

__global__ void edge_nri_fused(const float* __restrict__ node,
                               const float* __restrict__ edges,
                               const float* __restrict__ Wn2e, const float* __restrict__ bn2e,
                               const float* __restrict__ Wev1, const float* __restrict__ bev1,
                               const float* __restrict__ Wev2, const float* __restrict__ bev2,
                               const float* __restrict__ Wse1, const float* __restrict__ bse1,
                               const float* __restrict__ Wse2, const float* __restrict__ bse2,
                               float* __restrict__ outz, float* __restrict__ out2) {
    extern __shared__ char smem[];
    _Float16* wtN2E = (_Float16*)(smem + OFF_WN2E);
    _Float16* wtSE1 = (_Float16*)(smem + OFF_WSE1);
    _Float16* wtSE2 = (_Float16*)(smem + OFF_WSE2);
    _Float16* wtEV1 = (_Float16*)(smem + OFF_WEV1);
    float*    bias  = (float*)   (smem + OFF_BIAS);

    const int tid  = threadIdx.x;
    const int nthr = blockDim.x;

    // ---- stage weights transposed ([N][K]) as f16, once per block ----
    for (int idx = tid; idx < 256 * 128; idx += nthr) {      // W_n2e (256,128)
        int k = idx >> 7, n = idx & 127;
        wtN2E[n * 256 + k] = (_Float16)Wn2e[idx];
    }
    for (int idx = tid; idx < 128 * 64; idx += nthr) {       // W_se1 (128,64)
        int k = idx >> 6, n = idx & 63;
        wtSE1[n * 128 + k] = (_Float16)Wse1[idx];
    }
    for (int idx = tid; idx < 64 * 128; idx += nthr) {       // W_se2 (64,128)
        int k = idx >> 7, n = idx & 127;
        wtSE2[n * 64 + k] = (_Float16)Wse2[idx];
    }
    for (int idx = tid; idx < 128 * 64; idx += nthr) {       // W_e2v1 (128,64)
        int k = idx >> 6, n = idx & 63;
        wtEV1[n * 128 + k] = (_Float16)Wev1[idx];
    }
    for (int idx = tid; idx < 449; idx += nthr) {            // biases + W_e2v2
        float v;
        if      (idx < 128) v = bn2e[idx];
        else if (idx < 192) v = bse1[idx - 128];
        else if (idx < 320) v = bse2[idx - 192];
        else if (idx < 384) v = bev1[idx - 320];
        else if (idx < 448) v = Wev2[idx - 384];
        else                v = bev2[0];
        bias[idx] = v;
    }
    __syncthreads();

    const float* Bn2e = bias;
    const float* Bse1 = bias + 128;
    const float* Bse2 = bias + 192;
    const float* Bev1 = bias + 320;
    const float* Wv2  = bias + 384;
    const float  b2f  = bias[448];

    const int wv   = tid >> 5;
    const int lane = tid & 31;
    const int hi   = lane >> 4;     // half-wave select
    const int lm   = lane & 15;

    char* wbase = smem + PW_BASE + wv * PW_STRIDE;
    _Float16* S16 = (_Float16*)(wbase + PW_S16);
    _Float16* R16 = (_Float16*)(wbase + PW_R16);
    _Float16* ALN = (_Float16*)(wbase + PW_ALN);
    _Float16* AH  = (_Float16*)(wbase + PW_AH);
    _Float16* AZ  = (_Float16*)(wbase + PW_AZ);
    float*    EFN = (float*)   (wbase + PW_EFN);

    const int totalWaves = gridDim.x * (blockDim.x >> 5);
    const int gw = blockIdx.x * (blockDim.x >> 5) + wv;

    const int rr = lane >> 1;            // staging row (2 lanes per row)
    const int c0 = (lane & 1) * 64;      // staging column half

    for (int tile = gw; tile < NTILES; tile += totalWaves) {
        const int r0  = tile * 16;
        const int kb  = r0 >> 12;          // / (N*N)
        const int rem = r0 & 4095;
        const int ii  = rem >> 6;          // sender index (shared by tile)
        const int j0  = rem & 63;          // first receiver index

        // prefetch next tile's edges_input rows into cache hierarchy
        {
            const int tn = tile + totalWaves;
            if (tn < NTILES) {
                const float* p = edges + ((size_t)(tn * 16 + rr)) * DDIM + c0;
                __builtin_prefetch(p, 0, 1);
                __builtin_prefetch(p + 32, 0, 1);
            }
        }

        // ---- stage sender row (128 f32 -> f16), broadcast to all rows ----
        {
            const float4* sp = (const float4*)(node + ((size_t)(kb * 64 + ii)) * DDIM);
            *(v4h*)(S16 + lane * 4) = pack4(sp[lane]);
        }
        // ---- stage 16 receiver rows: 2 lanes/row, 64 floats each ----
        {
            const float4* rp = (const float4*)(node + ((size_t)(kb * 64 + j0 + rr)) * DDIM + c0);
            #pragma unroll
            for (int t = 0; t < 16; ++t)
                *(v4h*)(R16 + rr * 128 + c0 + t * 4) = pack4(rp[t]);
        }
        // ---- LayerNorm of edges_input rows -> ALN (f16) ----
        {
            const float4* ep = (const float4*)(edges + ((size_t)(r0 + rr)) * DDIM + c0);
            float4 x[16];
            float s = 0.f, ss = 0.f;
            #pragma unroll
            for (int t = 0; t < 16; ++t) {
                x[t] = ep[t];
                s  += x[t].x + x[t].y + x[t].z + x[t].w;
                ss += x[t].x * x[t].x + x[t].y * x[t].y + x[t].z * x[t].z + x[t].w * x[t].w;
            }
            s  += __shfl_xor(s, 1, 32);
            ss += __shfl_xor(ss, 1, 32);
            const float mu = s * (1.0f / 128.0f);
            const float rs = rsqrtf(ss * (1.0f / 128.0f) - mu * mu + 1e-5f);
            #pragma unroll
            for (int t = 0; t < 16; ++t) {
                float4 y; y.x = (x[t].x - mu) * rs; y.y = (x[t].y - mu) * rs;
                          y.z = (x[t].z - mu) * rs; y.w = (x[t].w - mu) * rs;
                *(v4h*)(ALN + rr * 128 + c0 + t * 4) = pack4(y);
            }
        }
        lds_fence();

        // ---- GEMM1: [sender|receiver](16x256) @ W_n2e -> gelu -> EFN (f32) ----
        {
            v16h a[8];
            #pragma unroll
            for (int kk = 0; kk < 4; ++kk) a[kk]     = ldsA(S16, kk * 32, hi);
            #pragma unroll
            for (int kk = 0; kk < 4; ++kk) a[kk + 4] = ldsA(R16 + lm * 128, kk * 32, hi);
            for (int t = 0; t < 8; ++t) {
                v16h b[8];
                #pragma unroll
                for (int kk = 0; kk < 8; ++kk)
                    b[kk] = ldsB(wtN2E, 256, t * 16 + lm, kk * 32, hi);
                v8f c = {};
                #pragma unroll
                for (int kk = 0; kk < 8; ++kk)
                    c = wmma16(a[kk], b[kk], c);
                const int col = t * 16 + lm;
                const float bb = Bn2e[col];
                #pragma unroll
                for (int r = 0; r < 8; ++r)
                    EFN[(r + hi * 8) * 128 + col] = gelu_f(c[r] + bb);
            }
        }
        // ---- GEMM2: e_ln(16x128) @ W_se1 -> relu -> AH (f16) ----
        {
            v16h a[4];
            #pragma unroll
            for (int kk = 0; kk < 4; ++kk) a[kk] = ldsA(ALN + lm * 128, kk * 32, hi);
            for (int t = 0; t < 4; ++t) {
                v16h b[4];
                #pragma unroll
                for (int kk = 0; kk < 4; ++kk)
                    b[kk] = ldsB(wtSE1, 128, t * 16 + lm, kk * 32, hi);
                v8f c = {};
                #pragma unroll
                for (int kk = 0; kk < 4; ++kk)
                    c = wmma16(a[kk], b[kk], c);
                const int col = t * 16 + lm;
                const float bb = Bse1[col];
                #pragma unroll
                for (int r = 0; r < 8; ++r)
                    AH[(r + hi * 8) * 64 + col] = (_Float16)fmaxf(c[r] + bb, 0.f);
            }
        }
        lds_fence();
        // ---- GEMM3: h(16x64) @ W_se2 ; z = c + b + EFN -> EFN (in place) + AZ ----
        {
            v16h a[2];
            #pragma unroll
            for (int kk = 0; kk < 2; ++kk) a[kk] = ldsA(AH + lm * 64, kk * 32, hi);
            for (int t = 0; t < 8; ++t) {
                v16h b[2];
                #pragma unroll
                for (int kk = 0; kk < 2; ++kk)
                    b[kk] = ldsB(wtSE2, 64, t * 16 + lm, kk * 32, hi);
                v8f c = {};
                #pragma unroll
                for (int kk = 0; kk < 2; ++kk)
                    c = wmma16(a[kk], b[kk], c);
                const int col = t * 16 + lm;
                const float bb = Bse2[col];
                #pragma unroll
                for (int r = 0; r < 8; ++r) {
                    const int row = r + hi * 8;
                    float z = c[r] + bb + EFN[row * 128 + col];
                    EFN[row * 128 + col] = z;           // z staged for coalesced store
                    AZ[row * 128 + col] = (_Float16)z;
                }
            }
        }
        lds_fence();
        // ---- coalesced b128 store of edges_z from EFN ----
        {
            float4* zp = (float4*)(outz + ((size_t)(r0 + rr)) * DDIM + c0);
            const float4* ep = (const float4*)(EFN + rr * 128 + c0);
            #pragma unroll
            for (int t = 0; t < 16; ++t)
                zp[t] = ep[t];
        }
        // ---- GEMM4: z(16x128) @ W_e2v1 -> relu -> dot W_e2v2 -> relu ----
        {
            float vacc[8];
            #pragma unroll
            for (int r = 0; r < 8; ++r) vacc[r] = 0.f;
            v16h a[4];
            #pragma unroll
            for (int kk = 0; kk < 4; ++kk) a[kk] = ldsA(AZ + lm * 128, kk * 32, hi);
            for (int t = 0; t < 4; ++t) {
                v16h b[4];
                #pragma unroll
                for (int kk = 0; kk < 4; ++kk)
                    b[kk] = ldsB(wtEV1, 128, t * 16 + lm, kk * 32, hi);
                v8f c = {};
                #pragma unroll
                for (int kk = 0; kk < 4; ++kk)
                    c = wmma16(a[kk], b[kk], c);
                const int col = t * 16 + lm;
                const float b1 = Bev1[col], w2 = Wv2[col];
                #pragma unroll
                for (int r = 0; r < 8; ++r)
                    vacc[r] += fmaxf(c[r] + b1, 0.f) * w2;
            }
            #pragma unroll
            for (int r = 0; r < 8; ++r) {
                vacc[r] += __shfl_xor(vacc[r], 8, 32);
                vacc[r] += __shfl_xor(vacc[r], 4, 32);
                vacc[r] += __shfl_xor(vacc[r], 2, 32);
                vacc[r] += __shfl_xor(vacc[r], 1, 32);
            }
            if (lm == 0) {
                #pragma unroll
                for (int r = 0; r < 8; ++r)
                    out2[(size_t)r0 + r + hi * 8] = fmaxf(vacc[r] + b2f, 0.f);
            }
        }
        lds_fence();   // protect per-wave buffers before next tile's staging
    }
}

extern "C" void kernel_launch(void* const* d_in, const int* in_sizes, int n_in,
                              void* d_out, int out_size, void* d_ws, size_t ws_size,
                              hipStream_t stream) {
    const float* node = (const float*)d_in[0];
    const float* edges = (const float*)d_in[1];
    const float* Wn2e = (const float*)d_in[2];
    const float* bn2e = (const float*)d_in[3];
    const float* Wev1 = (const float*)d_in[4];
    const float* bev1 = (const float*)d_in[5];
    const float* Wev2 = (const float*)d_in[6];
    const float* bev2 = (const float*)d_in[7];
    const float* Wse1 = (const float*)d_in[8];
    const float* bse1 = (const float*)d_in[9];
    const float* Wse2 = (const float*)d_in[10];
    const float* bse2 = (const float*)d_in[11];
    float* outz = (float*)d_out;
    float* out2 = outz + (size_t)NROWS * DDIM;

    hipFuncSetAttribute(reinterpret_cast<const void*>(edge_nri_fused),
                        hipFuncAttributeMaxDynamicSharedMemorySize, SMEM_BYTES);
    edge_nri_fused<<<1024, 256, SMEM_BYTES, stream>>>(
        node, edges, Wn2e, bn2e, Wev1, bev1, Wev2, bev2,
        Wse1, bse1, Wse2, bse2, outz, out2);
}
